// PixelEachSubstitutor_56427280335184
// MI455X (gfx1250) — matrix operand order repaired and checked by reading, not compiled, not run on hardware
//
#include <hip/hip_runtime.h>

// ---------------------------------------------------------------------------
// PixelEachSubstitutor, fully fused: one wave32 per batch element (pixel).
// Pass 1 (pes_pack_weights): repack every GEMM weight into f16 tiles laid out
//   exactly as the v_wmma B operand ([tile][lane][16 halves]) in d_ws.
// Pass 2 (pes_fused_kernel): per-pixel nested transformer; GEMMs are
//   v_wmma_f32_16x16x32_f16 with A tiles cached in registers (ds_load_b128
//   from tile-padded LDS) and B tiles as single 32B global loads (L2-hot).
// Softmax / LayerNorm / rank-1 (head_dim==1) attention are VALU.
// ---------------------------------------------------------------------------

typedef __attribute__((ext_vector_type(16))) _Float16 v16h;
typedef __attribute__((ext_vector_type(8)))  float    v8f;

union V16H { v16h v; _Float16 e[16]; };
union V8F  { v8f  v; float    e[8];  };

#define LN_EPS 1e-5f

// ----- packed-weight tile table (tile = 512 halves = 1KB) -------------------
constexpr int TS     = 512;
constexpr int T_PQKV = 3;   // NT=3, KT=1  (W: 33x11)
constexpr int T_PWO  = 1;   // NT=1, KT=1  (W: 11x11)
constexpr int T_PF1  = 2;   // NT=2, KT=1  (W: 32x11)
constexpr int T_PF2  = 1;   // NT=1, KT=1  (W: 11x32)
constexpr int T_LQKV = 20;  // NT=10,KT=2  (W: 147x49)
constexpr int T_LWO  = 8;   // NT=4, KT=2  (W: 49x49)
constexpr int T_CQKV = 2;   // NT=2, KT=1  (W: 30x10)
constexpr int T_CWO  = 1;   // NT=1, KT=1  (W: 10x10)
constexpr int O_PQKV = 0;
constexpr int O_PWO  = O_PQKV + 6 * T_PQKV;   // 18
constexpr int O_PF1  = O_PWO  + 6 * T_PWO;    // 24
constexpr int O_PF2  = O_PF1  + 6 * T_PF1;    // 36
constexpr int O_LQKV = O_PF2  + 6 * T_PF2;    // 42
constexpr int O_LWO  = O_LQKV + 6 * T_LQKV;   // 162
constexpr int O_CQKV = O_LWO  + 6 * T_LWO;    // 210
constexpr int O_CWO  = O_CQKV + T_CQKV;       // 212
constexpr int TOTAL_TILES = O_CWO + T_CWO;    // 213  (~213KB f16 in d_ws)

// ----- pass 1: weight repack into WMMA B-operand layout ---------------------
// B tile (32x16 KxN) packing used here (contiguous k, same as load_a_u): lane l<16: N=n0+l, K=k0+{0..15};
// lane l>=16: N=n0+l-16, K=k0+16+{0..15}. Stored as [tile][lane][16 halves].
__global__ __launch_bounds__(32) void pes_pack_weights(
    const float* __restrict__ pWin, const float* __restrict__ pWout,
    const float* __restrict__ pWf1, const float* __restrict__ pWf2,
    const float* __restrict__ LWin, const float* __restrict__ LWout,
    const float* __restrict__ CWin, const float* __restrict__ CWout,
    _Float16* __restrict__ wpk)
{
  const int t    = blockIdx.x;
  const int lane = threadIdx.x & 31;

  const float* W; int NW, KW, KT, rel;
  if (t < O_PWO)       { rel = t - O_PQKV; W = pWin  + (rel / T_PQKV) * 363;  NW = 33;  KW = 11; KT = 1; rel %= T_PQKV; }
  else if (t < O_PF1)  { rel = t - O_PWO;  W = pWout + (rel / T_PWO)  * 121;  NW = 11;  KW = 11; KT = 1; rel %= T_PWO; }
  else if (t < O_PF2)  { rel = t - O_PF1;  W = pWf1  + (rel / T_PF1)  * 352;  NW = 32;  KW = 11; KT = 1; rel %= T_PF1; }
  else if (t < O_LQKV) { rel = t - O_PF2;  W = pWf2  + (rel / T_PF2)  * 352;  NW = 11;  KW = 32; KT = 1; rel %= T_PF2; }
  else if (t < O_LWO)  { rel = t - O_LQKV; W = LWin  + (rel / T_LQKV) * 7203; NW = 147; KW = 49; KT = 2; rel %= T_LQKV; }
  else if (t < O_CQKV) { rel = t - O_LWO;  W = LWout + (rel / T_LWO)  * 2401; NW = 49;  KW = 49; KT = 2; rel %= T_LWO; }
  else if (t < O_CWO)  { rel = t - O_CQKV; W = CWin;                          NW = 30;  KW = 10; KT = 1; }
  else                 { rel = t - O_CWO;  W = CWout;                         NW = 10;  KW = 10; KT = 1; }

  const int kt = rel % KT, nt = rel / KT;
  const int n  = nt * 16 + (lane & 15);
  const int kb = kt * 32 + (((lane >> 4) & 1) << 4);
  const bool nv = (n < NW);
  const float* row = W + (nv ? n : 0) * KW;
  _Float16* dst = wpk + ((size_t)t * 32 + lane) * 16;
#pragma unroll
  for (int i = 0; i < 16; ++i) {
    const int k = kb + i;
    const bool valid = nv && (k < KW);
    const float f = row[(k < KW) ? k : 0];
    dst[i] = (_Float16)(valid ? f : 0.0f);
  }
}

// ----- pass 2 helpers -------------------------------------------------------

// Unconditional A-tile load (16x32 MxK) from tile-padded row-major f32 LDS.
// One contiguous 16-float run per lane (K=k0+16*(lane>=16)+{0..15}) -> ds_load_b128 x4.
__device__ __forceinline__ v16h load_a_u(const float* __restrict__ A, int ld,
                                         int m0, int k0, int lane) {
  V16H r;
  const int m   = m0 + (lane & 15);
  const int khi = ((lane >> 4) & 1) << 4;
  const float* row = A + m * ld + k0 + khi;
#pragma unroll
  for (int i = 0; i < 8; ++i) {
    const int kb = 2 * i;
#pragma unroll
    for (int t = 0; t < 2; ++t)
      r.e[2 * i + t] = (_Float16)row[kb + t];
  }
  return r.v;
}

// Unconditional D-tile store: VGPR r, lane l -> M=m0+r+(l>=16?8:0), N=n0+(l&15).
__device__ __forceinline__ void store_d_u(float* __restrict__ D, int ld,
                                          int m0, int n0, v8f acc, int lane) {
  V8F a; a.v = acc;
  const int n  = n0 + (lane & 15);
  const int mb = m0 + (((lane >> 4) & 1) << 3);
  float* col = D + mb * ld + n;
#pragma unroll
  for (int r = 0; r < 8; ++r) col[r * ld] = a.e[r];
}

// D[MxN] = A[MxK] @ W^T with W pre-packed as f16 B tiles. A tiles cached in
// registers across the N tiles; B tile = one aligned 32B load per lane.
template <int M, int N, int K, int LDA, int LDD>
__device__ __forceinline__ void gemm_pk(const float* __restrict__ A,
                                        const _Float16* __restrict__ Bpk,
                                        float* __restrict__ D, int lane) {
  constexpr int MT = (M + 15) / 16;
  constexpr int NT = (N + 15) / 16;
  constexpr int KT = (K + 31) / 32;
#pragma unroll
  for (int mt = 0; mt < MT; ++mt) {
    v16h at[KT];
#pragma unroll
    for (int kt = 0; kt < KT; ++kt) at[kt] = load_a_u(A, LDA, mt * 16, kt * 32, lane);
#pragma unroll
    for (int nt = 0; nt < NT; ++nt) {
      v8f acc = {};
#pragma unroll
      for (int kt = 0; kt < KT; ++kt) {
        const v16h b = *(const v16h*)(Bpk + (((size_t)(nt * KT + kt)) * 32 + lane) * 16);
        acc = __builtin_amdgcn_wmma_f32_16x16x32_f16(false, at[kt], false, b,
                                                     (short)0, acc, false, false);
      }
      store_d_u(D, LDD, mt * 16, nt * 16, acc, lane);
    }
  }
}

__global__ __launch_bounds__(32) void pes_fused_kernel(
    const float* __restrict__ x,
    const float* __restrict__ pln1,  const float* __restrict__ pln2,
    const float* __restrict__ LWf1,  const float* __restrict__ LWf2,
    const float* __restrict__ Lln1,  const float* __restrict__ Lln2,
    const float* __restrict__ CWf1,  const float* __restrict__ CWf2,
    const float* __restrict__ Cln1,  const float* __restrict__ Cln2,
    const float* __restrict__ Wdec,
    const _Float16* __restrict__ wpk,
    float* __restrict__ out)
{
  __shared__ float s_seq[49 * 12];     // original seq [49 x 11], ld 12
  __shared__ float s_z  [16 * 64];     // z_L [10 x 49], tile-padded, ld 64
  __shared__ float s_arena[8192];      // stage-local tile-padded buffers

  const int lane = threadIdx.x & 31;
  const int b  = blockIdx.x;
  const int n  = b / 900;
  const int hw = b % 900;
  const int h  = hw / 30;
  const int w  = hw % 30;

  // Stage-P arena layout
  float* s_x   = s_arena;              // [64 x 32]  activations, ld 32
  float* s_qkv = s_arena + 2048;       // [64 x 48]  qkv / proj out, ld 48
  float* s_h   = s_arena + 5120;       // [64 x 32]  attn-o / ffn hidden, ld 32

  for (int i = lane; i < 49 * 12; i += 32) s_seq[i] = 0.0f;
  for (int i = lane; i < 16 * 64; i += 32) s_z[i] = 0.0f;
  for (int i = lane; i < 8192; i += 32) s_arena[i] = 0.0f;
  __syncthreads();

  // ---------------- Stage 0: seq [49 x 11] (3x3 patch in 7x7 canvas) ----------
  for (int p = lane; p < 49; p += 32) {
    const int pi = p / 7, pj = p % 7;
    if (pi < 3 && pj < 3) {
      const int py = h + pi, px = w + pj;               // padded-canvas coords
      const bool border = (py == 0) || (py == 31) || (px == 0) || (px == 31);
      if (border) {
        s_seq[p * 12 + 10] = 1.0f;
      } else {
        const int iy = py - 1, ix = px - 1;
        for (int c = 0; c < 10; ++c)
          s_seq[p * 12 + c] = x[((n * 10 + c) * 30 + iy) * 30 + ix];
      }
    }
  }
  __syncthreads();
  for (int p = lane; p < 49; p += 32)
    for (int c = 0; c < 11; ++c)
      s_x[p * 32 + c] = s_seq[p * 12 + c];
  __syncthreads();

  // ---------------- Encoder P: 6 layers, d=11, S=49, nhead=11 (hd=1) ----------
  for (int lay = 0; lay < 6; ++lay) {
    const float* g1 = pln1 + lay * 11;
    const float* g2 = pln2 + lay * 11;

    // QKV: [49,11] @ [11,33] -> s_qkv (q:0-10 k:11-21 v:22-32)
    gemm_pk<49, 33, 11, 32, 48>(s_x, wpk + (size_t)(O_PQKV + lay * T_PQKV) * TS, s_qkv, lane);
    __syncthreads();

    // rank-1 attention; only the 9 in-window keys survive the -1e9 mask
    // (exp(-1e9 - mx) == 0 exactly in f32, identical to the reference).
    for (int hh = 0; hh < 11; ++hh) {
      for (int q = lane; q < 49; q += 32) {
        const float qv = s_qkv[q * 48 + hh];
        float mx = -3.4e38f;
#pragma unroll
        for (int ki = 0; ki < 9; ++ki) {
          const int k = (ki / 3) * 7 + (ki % 3);
          mx = fmaxf(mx, qv * s_qkv[k * 48 + 11 + hh]);
        }
        float se = 0.0f, ov = 0.0f;
#pragma unroll
        for (int ki = 0; ki < 9; ++ki) {
          const int k = (ki / 3) * 7 + (ki % 3);
          const float e = __expf(qv * s_qkv[k * 48 + 11 + hh] - mx);
          se += e;
          ov += e * s_qkv[k * 48 + 22 + hh];
        }
        s_h[q * 32 + hh] = ov / se;
      }
    }
    __syncthreads();

    // out-proj: [49,11] @ [11,11] -> s_qkv
    gemm_pk<49, 11, 11, 32, 48>(s_h, wpk + (size_t)(O_PWO + lay * T_PWO) * TS, s_qkv, lane);
    __syncthreads();

    // residual + LN1 -> s_x
    for (int r = lane; r < 49; r += 32) {
      float v[11]; float mu = 0.0f;
#pragma unroll
      for (int c = 0; c < 11; ++c) { v[c] = s_x[r * 32 + c] + s_qkv[r * 48 + c]; mu += v[c]; }
      mu *= (1.0f / 11.0f);
      float var = 0.0f;
#pragma unroll
      for (int c = 0; c < 11; ++c) { const float d = v[c] - mu; var += d * d; }
      var *= (1.0f / 11.0f);
      const float rs = rsqrtf(var + LN_EPS);
#pragma unroll
      for (int c = 0; c < 11; ++c) s_x[r * 32 + c] = (v[c] - mu) * rs * g1[c];
    }
    __syncthreads();

    // FFN: relu([49,11]@[11,32]) @ [32,11]
    gemm_pk<49, 32, 11, 32, 32>(s_x, wpk + (size_t)(O_PF1 + lay * T_PF1) * TS, s_h, lane);
    __syncthreads();
    for (int i = lane; i < 64 * 32; i += 32) s_h[i] = fmaxf(s_h[i], 0.0f);
    __syncthreads();
    gemm_pk<49, 11, 32, 32, 48>(s_h, wpk + (size_t)(O_PF2 + lay * T_PF2) * TS, s_qkv, lane);
    __syncthreads();

    // residual + LN2 -> s_x
    for (int r = lane; r < 49; r += 32) {
      float v[11]; float mu = 0.0f;
#pragma unroll
      for (int c = 0; c < 11; ++c) { v[c] = s_x[r * 32 + c] + s_qkv[r * 48 + c]; mu += v[c]; }
      mu *= (1.0f / 11.0f);
      float var = 0.0f;
#pragma unroll
      for (int c = 0; c < 11; ++c) { const float d = v[c] - mu; var += d * d; }
      var *= (1.0f / 11.0f);
      const float rs = rsqrtf(var + LN_EPS);
#pragma unroll
      for (int c = 0; c < 11; ++c) s_x[r * 32 + c] = (v[c] - mu) * rs * g2[c];
    }
    __syncthreads();
  }

  // ---------------- cp + z: z[c][p] = seq[p][c] + (border ? softmax_c(h) : 0) --
  for (int p = lane; p < 49; p += 32) {
    const bool m0 = (s_seq[p * 12 + 10] == 1.0f);
    float cp[10];
    if (m0) {
      float mx = -3.4e38f;
#pragma unroll
      for (int c = 0; c < 10; ++c) mx = fmaxf(mx, s_x[p * 32 + c]);
      float se = 0.0f;
#pragma unroll
      for (int c = 0; c < 10; ++c) { cp[c] = __expf(s_x[p * 32 + c] - mx); se += cp[c]; }
      const float inv = 1.0f / se;
#pragma unroll
      for (int c = 0; c < 10; ++c) cp[c] *= inv;
    } else {
#pragma unroll
      for (int c = 0; c < 10; ++c) cp[c] = 0.0f;
    }
    for (int c = 0; c < 10; ++c) s_z[c * 64 + p] = s_seq[p * 12 + c] + cp[c];
  }
  __syncthreads();

  // Stage-L arena layout
  float* s_qkvL = s_arena;             // [16 x 160] qkv, ld 160
  float* s_oL   = s_arena + 2560;      // [16 x 64]  attn-o, ld 64
  float* s_prL  = s_arena + 3584;      // [16 x 64]  proj out, ld 64

  // ---------------- Encoder L: 6 layers, d=49, S=10, nhead=49 (hd=1) ----------
  for (int lay = 0; lay < 6; ++lay) {
    const float* W1 = LWf1 + lay * 49;
    const float* W2 = LWf2 + lay * 49;
    const float* g1 = Lln1 + lay * 49;
    const float* g2 = Lln2 + lay * 49;

    // QKV: [10,49] @ [49,147] -> s_qkvL (q:0-48 k:49-97 v:98-146)
    gemm_pk<10, 147, 49, 64, 160>(s_z, wpk + (size_t)(O_LQKV + lay * T_LQKV) * TS, s_qkvL, lane);
    __syncthreads();

    // rank-1 attention over 49 heads, S = 10, no mask; o -> s_oL
    for (int hh = lane; hh < 49; hh += 32) {
      float qv[10], kv[10], vv[10];
#pragma unroll
      for (int s = 0; s < 10; ++s) {
        qv[s] = s_qkvL[s * 160 + hh];
        kv[s] = s_qkvL[s * 160 + 49 + hh];
        vv[s] = s_qkvL[s * 160 + 98 + hh];
      }
#pragma unroll
      for (int qi = 0; qi < 10; ++qi) {
        float mx = -3.4e38f;
#pragma unroll
        for (int k = 0; k < 10; ++k) mx = fmaxf(mx, qv[qi] * kv[k]);
        float se = 0.0f, ov = 0.0f;
#pragma unroll
        for (int k = 0; k < 10; ++k) {
          const float e = __expf(qv[qi] * kv[k] - mx);
          se += e; ov += e * vv[k];
        }
        s_oL[qi * 64 + hh] = ov / se;
      }
    }
    __syncthreads();

    // out-proj: [10,49] @ [49,49] -> s_prL
    gemm_pk<10, 49, 49, 64, 64>(s_oL, wpk + (size_t)(O_LWO + lay * T_LWO) * TS, s_prL, lane);
    __syncthreads();

    // residual + LN1 (rows = 10) -> s_z
    if (lane < 10) {
      const int s = lane;
      float mu = 0.0f;
      for (int c = 0; c < 49; ++c) {
        const float t = s_z[s * 64 + c] + s_prL[s * 64 + c];
        s_z[s * 64 + c] = t; mu += t;
      }
      mu *= (1.0f / 49.0f);
      float var = 0.0f;
      for (int c = 0; c < 49; ++c) { const float d = s_z[s * 64 + c] - mu; var += d * d; }
      var *= (1.0f / 49.0f);
      const float rs = rsqrtf(var + LN_EPS);
      for (int c = 0; c < 49; ++c) s_z[s * 64 + c] = (s_z[s * 64 + c] - mu) * rs * g1[c];
    }
    __syncthreads();

    // width-1 FFN + residual + LN2 -> s_z
    if (lane < 10) {
      const int s = lane;
      float dot = 0.0f;
      for (int c = 0; c < 49; ++c) dot += s_z[s * 64 + c] * W1[c];
      dot = fmaxf(dot, 0.0f);
      float mu = 0.0f;
      for (int c = 0; c < 49; ++c) {
        const float t = s_z[s * 64 + c] + dot * W2[c];
        s_z[s * 64 + c] = t; mu += t;
      }
      mu *= (1.0f / 49.0f);
      float var = 0.0f;
      for (int c = 0; c < 49; ++c) { const float d = s_z[s * 64 + c] - mu; var += d * d; }
      var *= (1.0f / 49.0f);
      const float rs = rsqrtf(var + LN_EPS);
      for (int c = 0; c < 49; ++c) s_z[s * 64 + c] = (s_z[s * 64 + c] - mu) * rs * g2[c];
    }
    __syncthreads();
  }

  // Stage-C arena layout
  float* s_zc   = s_arena;             // [64 x 32] z^T, ld 32
  float* s_qkvC = s_arena + 2048;      // [64 x 32] qkv, ld 32
  float* s_oC   = s_arena + 4096;      // [64 x 32] attn-o, ld 32
  float* s_prC  = s_arena + 6144;      // [64 x 32] proj out, ld 32

  // transpose z -> zc [49][10]
  for (int p = lane; p < 49; p += 32) {
#pragma unroll
    for (int c = 0; c < 10; ++c) s_zc[p * 32 + c] = s_z[c * 64 + p];
#pragma unroll
    for (int c = 10; c < 32; ++c) s_zc[p * 32 + c] = 0.0f;
  }
  __syncthreads();

  // ---------------- Encoder C: 1 layer, d=10, S=49, nhead=10 (hd=1) -----------
  {
    // QKV: [49,10] @ [10,30] -> s_qkvC (q:0-9 k:10-19 v:20-29)
    gemm_pk<49, 30, 10, 32, 32>(s_zc, wpk + (size_t)O_CQKV * TS, s_qkvC, lane);
    __syncthreads();

    for (int hh = 0; hh < 10; ++hh) {
      for (int q = lane; q < 49; q += 32) {
        const float qv = s_qkvC[q * 32 + hh];
        float mx = -3.4e38f;
        for (int k = 0; k < 49; ++k) mx = fmaxf(mx, qv * s_qkvC[k * 32 + 10 + hh]);
        float se = 0.0f, ov = 0.0f;
        for (int k = 0; k < 49; ++k) {
          const float e = __expf(qv * s_qkvC[k * 32 + 10 + hh] - mx);
          se += e; ov += e * s_qkvC[k * 32 + 20 + hh];
        }
        s_oC[q * 32 + hh] = ov / se;
      }
    }
    __syncthreads();

    // out-proj: [49,10] @ [10,10] -> s_prC
    gemm_pk<49, 10, 10, 32, 32>(s_oC, wpk + (size_t)O_CWO * TS, s_prC, lane);
    __syncthreads();

    // residual + LN1 -> s_zc
    for (int r = lane; r < 49; r += 32) {
      float v[10]; float mu = 0.0f;
#pragma unroll
      for (int c = 0; c < 10; ++c) { v[c] = s_zc[r * 32 + c] + s_prC[r * 32 + c]; mu += v[c]; }
      mu *= 0.1f;
      float var = 0.0f;
#pragma unroll
      for (int c = 0; c < 10; ++c) { const float d = v[c] - mu; var += d * d; }
      var *= 0.1f;
      const float rs = rsqrtf(var + LN_EPS);
#pragma unroll
      for (int c = 0; c < 10; ++c) s_zc[r * 32 + c] = (v[c] - mu) * rs * Cln1[c];
    }
    __syncthreads();

    // width-1 FFN + residual + LN2 -> s_zc
    for (int r = lane; r < 49; r += 32) {
      float dot = 0.0f;
#pragma unroll
      for (int c = 0; c < 10; ++c) dot += s_zc[r * 32 + c] * CWf1[c];
      dot = fmaxf(dot, 0.0f);
      float v[10]; float mu = 0.0f;
#pragma unroll
      for (int c = 0; c < 10; ++c) { v[c] = s_zc[r * 32 + c] + dot * CWf2[c]; mu += v[c]; }
      mu *= 0.1f;
      float var = 0.0f;
#pragma unroll
      for (int c = 0; c < 10; ++c) { const float d = v[c] - mu; var += d * d; }
      var *= 0.1f;
      const float rs = rsqrtf(var + LN_EPS);
#pragma unroll
      for (int c = 0; c < 10; ++c) s_zc[r * 32 + c] = (v[c] - mu) * rs * Cln2[c];
    }
    __syncthreads();
  }

  // ---------------- Decoder: out[n,c,h,w] = sum_p zfinal[c,p] * Wdec[p] -------
  if (lane < 10) {
    const int c = lane;
    float acc = 0.0f;
    for (int p = 0; p < 49; ++p) acc += s_zc[p * 32 + c] * Wdec[p];
    out[((n * 10 + c) * 30 + h) * 30 + w] = acc;
  }
}

extern "C" void kernel_launch(void* const* d_in, const int* in_sizes, int n_in,
                              void* d_out, int out_size, void* d_ws, size_t ws_size,
                              hipStream_t stream) {
  (void)n_in; (void)out_size; (void)ws_size;
  const float* x    = (const float*)d_in[0];
  const float* pWin = (const float*)d_in[1];
  const float* pWout= (const float*)d_in[2];
  const float* pWf1 = (const float*)d_in[3];
  const float* pWf2 = (const float*)d_in[4];
  const float* pln1 = (const float*)d_in[5];
  const float* pln2 = (const float*)d_in[6];
  const float* LWin = (const float*)d_in[7];
  const float* LWout= (const float*)d_in[8];
  const float* LWf1 = (const float*)d_in[9];
  const float* LWf2 = (const float*)d_in[10];
  const float* Lln1 = (const float*)d_in[11];
  const float* Lln2 = (const float*)d_in[12];
  const float* CWin = (const float*)d_in[13];
  const float* CWout= (const float*)d_in[14];
  const float* CWf1 = (const float*)d_in[15];
  const float* CWf2 = (const float*)d_in[16];
  const float* Cln1 = (const float*)d_in[17];
  const float* Cln2 = (const float*)d_in[18];
  const float* Wdec = (const float*)d_in[19];
  float* out = (float*)d_out;
  _Float16* wpk = (_Float16*)d_ws;     // 213 tiles * 1KB = ~213KB scratch

  // pass 1: repack weights into WMMA B-operand f16 tiles (deterministic)
  pes_pack_weights<<<dim3(TOTAL_TILES), dim3(32), 0, stream>>>(
      pWin, pWout, pWf1, pWf2, LWin, LWout, CWin, CWout, wpk);

  // pass 2: fused per-pixel nested transformer, one wave32 per pixel
  const int N = in_sizes[0] / (10 * 30 * 30);   // = 2
  const int B = N * 30 * 30;
  pes_fused_kernel<<<dim3(B), dim3(32), 0, stream>>>(
      x, pln1, pln2, LWf1, LWf2, Lln1, Lln2,
      CWf1, CWf2, Cln1, Cln2, Wdec, wpk, out);
}
